// AttnBlock_39505109188654
// MI455X (gfx1250) — matrix operand order repaired and checked by reading, not compiled, not run on hardware
//
#include <hip/hip_runtime.h>
#include <hip/hip_bf16.h>
#include <hip/hip_fp16.h>

typedef __attribute__((ext_vector_type(16))) _Float16 v16h;
typedef __attribute__((ext_vector_type(8)))  _Float16 v8h;
typedef __attribute__((ext_vector_type(8)))  float    v8f;

#define NB 4      // batch
#define C  256    // channels
#define NG 8      // groups
#define CG 32     // channels per group
#define NP 4096   // pixels (H*W)
#define EPS 1e-5f

// ---------------------------------------------------------------------------
// WMMA helpers (CDNA5 16x16x32 f16 -> f32). Fragment layouts per ISA 7.12.2:
//  A (16Mx32K f16): lane<16 holds M=lane, K in {0..7,16..23};
//                   lane>=16 holds M=lane-16, K in {8..15,24..31}
//  B (32Kx16N f16): lane holds N=lane%16, K same as A (8*(lane/16)+{0..7,16..23})
//  D (16x16 f32):   VGPR r, lane L -> (M = r + 8*(L/16), N = L%16)
// ---------------------------------------------------------------------------
__device__ __forceinline__ v16h make_a(v8h lo, v8h hi) {
  v16h r;
#pragma unroll
  for (int t = 0; t < 8; ++t) { r[t] = lo[t]; r[t + 8] = hi[t]; }
  return r;
}

// A-fragment from row-major f16 matrix [rows, ld] at (row0, k0), k-chunk of 32
__device__ __forceinline__ v16h load_a_g(const _Float16* base, int ld, int row0, int k0) {
  int lane = threadIdx.x & 31;
  int lr = lane & 15, hf = lane >> 4;
  const _Float16* p = base + (size_t)(row0 + lr) * ld + k0 + 8 * hf;
  return make_a(*(const v8h*)p, *(const v8h*)(p + 16));
}

// B-fragment: element (k, n) lives at base[(n0+ln)*ld + k0 + k] (k contiguous)
__device__ __forceinline__ v16h load_b_g(const _Float16* base, int ld, int k0, int n0) {
  int lane = threadIdx.x & 31;
  int ln = lane & 15, hf = lane >> 4;
  const _Float16* p = base + (size_t)(n0 + ln) * ld + k0 + 8 * hf;
  return make_a(*(const v8h*)p, *(const v8h*)(p + 16));
}

// B-fragment from an LDS tile: element (k, n) at base[(n0+ln)*stride + k0 + k]
__device__ __forceinline__ v16h load_b_lds(const _Float16* base, int stride_h,
                                           int k0, int n0) {
  int lane = threadIdx.x & 31;
  int ln = lane & 15, hf = lane >> 4;
  const _Float16* p = base + (n0 + ln) * stride_h + k0 + 8 * hf;
  return make_a(*(const v8h*)p, *(const v8h*)(p + 16));
}

__device__ __forceinline__ v8f wmma16(v16h a, v16h b, v8f c) {
  return __builtin_amdgcn_wmma_f32_16x16x32_f16(
      /*neg_a=*/false, a, /*neg_b=*/false, b,
      /*c_mod=*/(short)0, c, /*reuse_a=*/false, /*reuse_b=*/false);
}

// Async global->LDS copy, 16B per lane, tracked with ASYNCcnt (ISA 08 §4).
__device__ __forceinline__ void async_ld_b128(unsigned lds_off, const void* gaddr) {
  asm volatile("global_load_async_to_lds_b128 %0, %1, off"
               :: "v"(lds_off), "v"(gaddr) : "memory");
}
__device__ __forceinline__ void wait_async(int n) {
  if (n == 0)
    asm volatile("s_wait_asynccnt 0" ::: "memory");
  else
    asm volatile("s_wait_asynccnt 8" ::: "memory");
}

// ---------------------------------------------------------------------------
// Kernel 1: GroupNorm statistics. One block per (b, g); 32*4096 elems each.
// ---------------------------------------------------------------------------
__global__ __launch_bounds__(256) void gn_stats(const float* __restrict__ x,
                                                float* __restrict__ mean,
                                                float* __restrict__ rstd) {
  const int bg = blockIdx.x;
  const int bb = bg / NG, g = bg % NG;
  const float* p = x + ((size_t)bb * C + g * CG) * NP;
  const int total = CG * NP;
  float s = 0.f, ss = 0.f;
  for (int i = threadIdx.x; i < total; i += 256) {
    float v = p[i];
    s += v; ss += v * v;
  }
  __shared__ float rs_[256], rss_[256];
  rs_[threadIdx.x] = s; rss_[threadIdx.x] = ss;
  __syncthreads();
  for (int st = 128; st > 0; st >>= 1) {
    if (threadIdx.x < st) {
      rs_[threadIdx.x] += rs_[threadIdx.x + st];
      rss_[threadIdx.x] += rss_[threadIdx.x + st];
    }
    __syncthreads();
  }
  if (threadIdx.x == 0) {
    float m = rs_[0] / total;
    float var = rss_[0] / total - m * m;
    mean[bg] = m;
    rstd[bg] = rsqrtf(var + EPS);
  }
}

// ---------------------------------------------------------------------------
// Kernel 2: weight f32 -> f16 convert
// ---------------------------------------------------------------------------
__global__ void cvt_f16(const float* __restrict__ s, _Float16* __restrict__ d, int n) {
  int i = blockIdx.x * blockDim.x + threadIdx.x;
  if (i < n) d[i] = (_Float16)s[i];
}

// ---------------------------------------------------------------------------
// Kernel 3: apply GroupNorm affine and transpose [b,c,n] f32 -> [b,n,c] f16
// via a 64x64 LDS tile (coalesced read over i, 32B-vector writes over c).
// ---------------------------------------------------------------------------
__global__ __launch_bounds__(256) void gn_apply_t(const float* __restrict__ x,
                                                  const float* __restrict__ gsc,
                                                  const float* __restrict__ gbi,
                                                  const float* __restrict__ mean,
                                                  const float* __restrict__ rstd,
                                                  _Float16* __restrict__ hn) {
  __shared__ _Float16 tile[64][65];
  const int b = blockIdx.z;
  const int c0 = blockIdx.y * 64;
  const int i0 = blockIdx.x * 64;
  const int t = threadIdx.x;
  {
    int cl = t >> 2;
    int ic = (t & 3) * 16;
    int c = c0 + cl;
    int g = c / CG;
    float m = mean[b * NG + g], r = rstd[b * NG + g];
    float sc = gsc[c] * r;
    float bi = gbi[c] - m * sc;  // (x-m)*r*s + b == x*sc + bi
    const float* xp = x + ((size_t)b * C + c) * NP + i0 + ic;
#pragma unroll
    for (int k = 0; k < 16; ++k)
      tile[cl][ic + k] = (_Float16)(xp[k] * sc + bi);
  }
  __syncthreads();
  {
    int il = t >> 2;
    int cc = (t & 3) * 16;
    _Float16* dp = hn + ((size_t)b * NP + i0 + il) * C + c0 + cc;
    v8h lo, hi;
#pragma unroll
    for (int k = 0; k < 8; ++k) {
      lo[k] = tile[cc + k][il];
      hi[k] = tile[cc + 8 + k][il];
    }
    *(v8h*)dp = lo;
    *(v8h*)(dp + 8) = hi;
  }
}

// ---------------------------------------------------------------------------
// Kernel 4: fused QKV GEMM. q[i,co]=sum_c hn[i,c]*wq[co,c] (+bias), ditto k,v.
// Per wave: 16(M=i) x 64(N=co) for all three projections, A reused 12x.
// Q,K stored [n,c] f16; V stored [c,n] f16 (transposed store for the PV GEMM).
// ---------------------------------------------------------------------------
__global__ __launch_bounds__(256) void qkv_gemm(const _Float16* __restrict__ hn,
                                                const _Float16* __restrict__ wq16,
                                                const _Float16* __restrict__ wk16,
                                                const _Float16* __restrict__ wv16,
                                                const float* __restrict__ bq,
                                                const float* __restrict__ bk,
                                                const float* __restrict__ bv,
                                                _Float16* __restrict__ qm,
                                                _Float16* __restrict__ km,
                                                _Float16* __restrict__ vm) {
  const int b = blockIdx.y;
  const int w = threadIdx.x >> 5;
  const int lane = threadIdx.x & 31;
  const int ln = lane & 15, hf = lane >> 4;
  const int mi = blockIdx.x * 32 + (w >> 2) * 16;
  const int n0 = (w & 3) * 64;
  const _Float16* hb = hn + (size_t)b * NP * C;

  v8f zero = {};
  v8f aq[4], ak[4], av[4];
#pragma unroll
  for (int t = 0; t < 4; ++t) { aq[t] = zero; ak[t] = zero; av[t] = zero; }

  for (int c0 = 0; c0 < C; c0 += 32) {
    v16h a = load_a_g(hb, C, mi, c0);
#pragma unroll
    for (int nt = 0; nt < 4; ++nt) {
      aq[nt] = wmma16(a, load_b_g(wq16, C, c0, n0 + nt * 16), aq[nt]);
      ak[nt] = wmma16(a, load_b_g(wk16, C, c0, n0 + nt * 16), ak[nt]);
      av[nt] = wmma16(a, load_b_g(wv16, C, c0, n0 + nt * 16), av[nt]);
    }
  }
  _Float16* qb = qm + (size_t)b * NP * C;
  _Float16* kb = km + (size_t)b * NP * C;
  _Float16* vb = vm + (size_t)b * C * NP;
#pragma unroll
  for (int nt = 0; nt < 4; ++nt) {
    int co = n0 + nt * 16 + ln;
    float bqv = bq[co], bkv = bk[co], bvv = bv[co];
#pragma unroll
    for (int r = 0; r < 8; ++r) {
      int i = mi + r + 8 * hf;
      qb[(size_t)i * C + co] = (_Float16)(aq[nt][r] + bqv);
      kb[(size_t)i * C + co] = (_Float16)(ak[nt][r] + bkv);
      vb[(size_t)co * NP + i] = (_Float16)(av[nt][r] + bvv);
    }
  }
}

// ---------------------------------------------------------------------------
// Kernel 5: flash attention. Block = 4 waves over 64 query rows; K/V tiles
// for each 32-key step are staged ONCE per block into LDS with async
// global->LDS copies (ASYNCcnt), with the V transfer overlapped behind the
// S = Q*K^T WMMAs + online softmax. B-fragments are fed from LDS with
// conflict-free padded strides.
//   K tile: 32 rows (keys) x 256 ch, stride 264 halves (132 dw -> +4 banks/row)
//   V tile: 256 rows (ch) x 32 keys, stride 40 halves (20 dw -> distinct banks)
// ---------------------------------------------------------------------------
#define KSTR 264   // halves per K-tile row (512B data + 16B pad)
#define VSTR 40    // halves per V-tile row (64B data + 16B pad)

__global__ __launch_bounds__(128) void attn_flash(const _Float16* __restrict__ qm,
                                                  const _Float16* __restrict__ km,
                                                  const _Float16* __restrict__ vm,
                                                  _Float16* __restrict__ ao) {
  const int b = blockIdx.y;
  const int w = threadIdx.x >> 5;
  const int lane = threadIdx.x & 31;
  const int ln = lane & 15, hf = lane >> 4;
  const int i0 = blockIdx.x * 64 + w * 16;

  const _Float16* qb = qm + (size_t)b * NP * C;
  const _Float16* kb = km + (size_t)b * NP * C;
  const _Float16* vb = vm + (size_t)b * C * NP;

  __shared__ __align__(16) _Float16 ktile[32][KSTR];
  __shared__ __align__(16) _Float16 vtile[256][VSTR];
  __shared__ __align__(16) _Float16 pbuf[4][16][48];

  const unsigned kbase = (unsigned)(size_t)&ktile[0][0];
  const unsigned vbase = (unsigned)(size_t)&vtile[0][0];

  v16h qf[8];
#pragma unroll
  for (int cc = 0; cc < 8; ++cc) qf[cc] = load_a_g(qb, C, i0, cc * 32);

  v8f zero = {};
  v8f o[16];
#pragma unroll
  for (int t = 0; t < 16; ++t) o[t] = zero;
  float mrow[8], lrow[8];
#pragma unroll
  for (int r = 0; r < 8; ++r) { mrow[r] = -3.0e38f; lrow[r] = 0.0f; }

  const int tid = threadIdx.x;

  for (int j0 = 0; j0 < NP; j0 += 32) {
    // ---- stage K tile (16 KB), then V tile (16 KB): 16B per lane per issue
#pragma unroll
    for (int q = 0; q < 8; ++q) {
      int e = tid + 128 * q;          // 0..1023 16B-chunks
      int jl = e >> 5, ch = e & 31;   // row, 16B chunk within row
      async_ld_b128(kbase + (unsigned)(jl * KSTR * 2 + ch * 16),
                    kb + (size_t)(j0 + jl) * C + ch * 8);
    }
#pragma unroll
    for (int q = 0; q < 8; ++q) {
      int e = tid + 128 * q;          // 0..1023
      int cr = e >> 2, ch = e & 3;    // channel row, 16B chunk (8 keys)
      async_ld_b128(vbase + (unsigned)(cr * VSTR * 2 + ch * 16),
                    vb + (size_t)cr * NP + j0 + ch * 8);
    }
    wait_async(8);        // K tile landed (V may still be in flight)
    __syncthreads();

    // ---- S tile: 16 queries x 32 keys from LDS K tile
    v8f s0 = zero, s1 = zero;
#pragma unroll
    for (int cc = 0; cc < 8; ++cc) {
      s0 = wmma16(qf[cc], load_b_lds(&ktile[0][0], KSTR, cc * 32, 0), s0);
      s1 = wmma16(qf[cc], load_b_lds(&ktile[0][0], KSTR, cc * 32, 16), s1);
    }
    // ---- online softmax (row max/sum via 16-lane shuffles)
    const float sc = 0.0625f;  // 1/sqrt(256)
    float av[8];
#pragma unroll
    for (int r = 0; r < 8; ++r) {
      float a = s0[r] * sc, bbv = s1[r] * sc;
      float t = fmaxf(a, bbv);
#pragma unroll
      for (int off = 1; off < 16; off <<= 1)
        t = fmaxf(t, __shfl_xor(t, off, 32));
      float nm = fmaxf(mrow[r], t);
      float al = __expf(mrow[r] - nm);
      mrow[r] = nm;
      float p0 = __expf(a - nm), p1 = __expf(bbv - nm);
      s0[r] = p0; s1[r] = p1;
      float rs = p0 + p1;
#pragma unroll
      for (int off = 1; off < 16; off <<= 1)
        rs += __shfl_xor(rs, off, 32);
      lrow[r] = lrow[r] * al + rs;
      av[r] = al;
    }
#pragma unroll
    for (int t = 0; t < 16; ++t)
#pragma unroll
      for (int r = 0; r < 8; ++r) o[t][r] *= av[r];

    // ---- D-layout P -> A-fragment via per-wave LDS staging
#pragma unroll
    for (int r = 0; r < 8; ++r) {
      pbuf[w][r + 8 * hf][ln]      = (_Float16)s0[r];
      pbuf[w][r + 8 * hf][ln + 16] = (_Float16)s1[r];
    }
    wait_async(0);        // V tile landed
    __syncthreads();
    const _Float16* pp = &pbuf[w][ln][8 * hf];
    v16h ap = make_a(*(const v8h*)pp, *(const v8h*)(pp + 16));

    // ---- O(16x256) += P(16x32) * V(32x256) from LDS V tile
#pragma unroll
    for (int ct = 0; ct < 16; ++ct)
      o[ct] = wmma16(ap, load_b_lds(&vtile[0][0], VSTR, 0, ct * 16), o[ct]);

    __syncthreads();      // WAR guard before next iteration's async fills
  }

  _Float16* aob = ao + (size_t)b * NP * C;
  float inv[8];
#pragma unroll
  for (int r = 0; r < 8; ++r) inv[r] = 1.0f / lrow[r];
#pragma unroll
  for (int ct = 0; ct < 16; ++ct)
#pragma unroll
    for (int r = 0; r < 8; ++r)
      aob[(size_t)(i0 + r + 8 * hf) * C + ct * 16 + ln] =
          (_Float16)(o[ct][r] * inv[r]);
}

// ---------------------------------------------------------------------------
// Kernel 6: projection GEMM + bias + residual. out[b,co,i] = x + wp@ao + bp
// ---------------------------------------------------------------------------
__global__ __launch_bounds__(256) void proj_residual(const float* __restrict__ x,
                                                     const _Float16* __restrict__ ao,
                                                     const _Float16* __restrict__ wp16,
                                                     const float* __restrict__ bp,
                                                     float* __restrict__ out) {
  const int b = blockIdx.y;
  const int w = threadIdx.x >> 5;
  const int lane = threadIdx.x & 31;
  const int ln = lane & 15, hf = lane >> 4;
  const int mi = blockIdx.x * 32 + (w >> 2) * 16;
  const int n0 = (w & 3) * 64;
  const _Float16* ab = ao + (size_t)b * NP * C;

  v8f zero = {};
  v8f acc[4];
#pragma unroll
  for (int t = 0; t < 4; ++t) acc[t] = zero;
  for (int c0 = 0; c0 < C; c0 += 32) {
    v16h a = load_a_g(ab, C, mi, c0);
#pragma unroll
    for (int nt = 0; nt < 4; ++nt)
      acc[nt] = wmma16(a, load_b_g(wp16, C, c0, n0 + nt * 16), acc[nt]);
  }
#pragma unroll
  for (int nt = 0; nt < 4; ++nt) {
    int co = n0 + nt * 16 + ln;
    float bpv = bp[co];
#pragma unroll
    for (int r = 0; r < 8; ++r) {
      int i = mi + r + 8 * hf;
      size_t idx = ((size_t)b * C + co) * NP + i;
      out[idx] = x[idx] + bpv + acc[nt][r];
    }
  }
}

// ---------------------------------------------------------------------------
// Launch
// ---------------------------------------------------------------------------
extern "C" void kernel_launch(void* const* d_in, const int* in_sizes, int n_in,
                              void* d_out, int out_size, void* d_ws, size_t ws_size,
                              hipStream_t stream) {
  const float* x   = (const float*)d_in[0];
  const float* gns = (const float*)d_in[1];
  const float* gnb = (const float*)d_in[2];
  const float* wq  = (const float*)d_in[3];
  const float* bq  = (const float*)d_in[4];
  const float* wk  = (const float*)d_in[5];
  const float* bk  = (const float*)d_in[6];
  const float* wv  = (const float*)d_in[7];
  const float* bv  = (const float*)d_in[8];
  const float* wp  = (const float*)d_in[9];
  const float* bp  = (const float*)d_in[10];
  float* out = (float*)d_out;

  // Workspace layout (~42.5 MB): stats, f16 weights, hn_t, q, k, v, attn_out
  float* mean = (float*)d_ws;                 // 32
  float* rstd = mean + 32;                    // 32
  _Float16* w16  = (_Float16*)(mean + 64);    // 256B offset, 16B aligned
  _Float16* wq16 = w16;
  _Float16* wk16 = w16 + (size_t)C * C;
  _Float16* wv16 = w16 + 2 * (size_t)C * C;
  _Float16* wp16 = w16 + 3 * (size_t)C * C;
  _Float16* hn = w16 + 4 * (size_t)C * C;
  const size_t mat = (size_t)NB * NP * C;
  _Float16* qm = hn + mat;
  _Float16* km = qm + mat;
  _Float16* vm = km + mat;
  _Float16* ao = vm + mat;

  gn_stats<<<dim3(NB * NG), dim3(256), 0, stream>>>(x, mean, rstd);
  cvt_f16<<<dim3(C * C / 256), dim3(256), 0, stream>>>(wq, wq16, C * C);
  cvt_f16<<<dim3(C * C / 256), dim3(256), 0, stream>>>(wk, wk16, C * C);
  cvt_f16<<<dim3(C * C / 256), dim3(256), 0, stream>>>(wv, wv16, C * C);
  cvt_f16<<<dim3(C * C / 256), dim3(256), 0, stream>>>(wp, wp16, C * C);
  gn_apply_t<<<dim3(NP / 64, C / 64, NB), dim3(256), 0, stream>>>(
      x, gns, gnb, mean, rstd, hn);
  qkv_gemm<<<dim3(NP / 32, NB), dim3(256), 0, stream>>>(
      hn, wq16, wk16, wv16, bq, bk, bv, qm, km, vm);
  attn_flash<<<dim3(NP / 64, NB), dim3(128), 0, stream>>>(qm, km, vm, ao);
  proj_residual<<<dim3(NP / 32, NB), dim3(256), 0, stream>>>(x, ao, wp16, bp, out);
}
